// GroupProjection_49976239456835
// MI455X (gfx1250) — matrix operand order repaired and checked by reading, not compiled, not run on hardware
//
#include <hip/hip_runtime.h>

// ---------- problem constants (from reference) ----------
#define DDIM   512
#define NROWS  8192
#define BATCH  8
#define NPROJ  4
#define NGRP   512
#define GSZ    16
#define KSTEPS 16      // 512 / 32 (K per bf16 WMMA)
#define CTILES 32      // 512 / 16 columns
#define CT_PER_WAVE 8  // 4 waves * 8 = 32 column tiles
#define MT     4       // batches per block: 4 M-tiles share each B fragment
#define APAD   8       // LDS row padding (bf16 elems) -> conflict-free frag reads

typedef __attribute__((ext_vector_type(16))) __bf16  v16bf;
typedef __attribute__((ext_vector_type(8)))  float   v8f;
typedef __attribute__((ext_vector_type(4)))  float   f4;
typedef __attribute__((ext_vector_type(4)))  unsigned int u4;
typedef __attribute__((ext_vector_type(2)))  unsigned int u2;

// round-to-nearest-even f32 -> bf16 (bit-level)
__device__ __forceinline__ unsigned short bf16_rn(float f) {
    unsigned u = __float_as_uint(f);
    unsigned r = u + 0x7FFFu + ((u >> 16) & 1u);
    return (unsigned short)(r >> 16);
}
__device__ __forceinline__ float bf16_f32(unsigned short h) {
    return __uint_as_float(((unsigned)h) << 16);
}
__device__ __forceinline__ unsigned pack2(unsigned short a, unsigned short b) {
    return (unsigned)a | ((unsigned)b << 16);
}

// ---------------------------------------------------------------------------
// Pack W[i] (f32, DxD row-major, used as B) into bf16 hi/lo fragments in
// per-lane WMMA B-operand order:
//   frag(i,kt,ct)[lane][e] = W[i][ kt*32 + (lane>>4)*16 + e ][ ct*16 + (lane&15) ]
// ---------------------------------------------------------------------------
__global__ void pack_w_kernel(const float* __restrict__ W,
                              unsigned short* __restrict__ whi,
                              unsigned short* __restrict__ wlo) {
    int t = blockIdx.x * blockDim.x + threadIdx.x;   // NPROJ*16*32*32 threads
    int lane =  t        & 31;
    int ct   = (t >> 5)  & 31;
    int kt   = (t >> 10) & 15;
    int i    =  t >> 14;
    int n     = ct * 16 + (lane & 15);
    int kbase = kt * 32 + (lane >> 4) * 16;
    size_t off = (size_t)t * 16;
    #pragma unroll
    for (int e = 0; e < 16; ++e) {
        float w = W[((size_t)i * DDIM + (kbase + e)) * DDIM + n];
        unsigned short hi = bf16_rn(w);
        unsigned short lo = bf16_rn(w - bf16_f32(hi));
        whi[off + e] = hi;
        wlo[off + e] = lo;
    }
}

// ---------------------------------------------------------------------------
// Per iteration: next = cur (f32 copy) AND emit the bf16 hi/lo split of
// (scale * cur) so the GEMM inner loop does no conversion VALU at all.
// ---------------------------------------------------------------------------
__global__ void split_copy_kernel(const f4* __restrict__ src, f4* __restrict__ dst,
                                  unsigned short* __restrict__ hi,
                                  unsigned short* __restrict__ lo,
                                  float scale, int n4) {
    int t = blockIdx.x * blockDim.x + threadIdx.x;
    if (t >= n4) return;
    f4 v = src[t];
    dst[t] = v;
    f4 s = v * scale;
    unsigned short h0 = bf16_rn(s[0]), h1 = bf16_rn(s[1]);
    unsigned short h2 = bf16_rn(s[2]), h3 = bf16_rn(s[3]);
    unsigned short l0 = bf16_rn(s[0] - bf16_f32(h0));
    unsigned short l1 = bf16_rn(s[1] - bf16_f32(h1));
    unsigned short l2 = bf16_rn(s[2] - bf16_f32(h2));
    unsigned short l3 = bf16_rn(s[3] - bf16_f32(h3));
    u2 ph; ph[0] = pack2(h0, h1); ph[1] = pack2(h2, h3);
    u2 pl; pl[0] = pack2(l0, l1); pl[1] = pack2(l2, l3);
    *(u2*)(hi + (size_t)t * 4) = ph;
    *(u2*)(lo + (size_t)t * 4) = pl;
}

// ---------------------------------------------------------------------------
// One block = (proj i, group g, 4 batches b0..b0+3): gather 4x16 pre-split
// rows (same gather indices, four batches), run the bf16x3 split GEMM with
// each B fragment feeding FOUR M-tiles (4x B reuse), scatter-add into next.
// ---------------------------------------------------------------------------
__global__ __launch_bounds__(128)
void group_gemm_kernel(const unsigned short* __restrict__ xhi,
                       const unsigned short* __restrict__ xlo,
                       float* __restrict__ next,
                       const unsigned short* __restrict__ whi,
                       const unsigned short* __restrict__ wlo,
                       const int* __restrict__ groups) {
    __shared__ __align__(16) unsigned short AsHi[MT][GSZ][DDIM + APAD];
    __shared__ __align__(16) unsigned short AsLo[MT][GSZ][DDIM + APAD];
    __shared__ int sIdx[GSZ];

    const int g  = blockIdx.x;
    const int b0 = blockIdx.y * MT;
    const int i  = blockIdx.z;
    const int t = threadIdx.x;
    const int lane = t & 31;
    const int wave = t >> 5;
    const int m    = lane & 15;   // A row in tile / C column in tile
    const int hsel = lane >> 4;   // lane-half selector per ISA layouts

    if (t < GSZ) sIdx[t] = groups[(((size_t)i * NGRP) + g) * GSZ + t];
    __syncthreads();

    // ---- stage gathered bf16 hi/lo rows (4 x 16 x 512 each) into LDS ----
    {
        const int row = t >> 1;          // 64 rows, 2 threads per row
        const int mt  = row >> 4;        // which batch of the quad
        const int r   = row & 15;        // group row
        const int seg = t & 1;           // 256 bf16 per (row,seg)
        const size_t rbase = ((size_t)(b0 + mt) * NROWS + sIdx[r]) * DDIM + seg * 256;
        const u4* sh = (const u4*)(xhi + rbase);
        const u4* sl = (const u4*)(xlo + rbase);
        #pragma unroll
        for (int j = 0; j < 32; ++j) {
            *(u4*)&AsHi[mt][r][seg * 256 + j * 8] = sh[j];
            *(u4*)&AsLo[mt][r][seg * 256 + j * 8] = sl[j];
        }
    }
    __syncthreads();

    v8f acc[MT][CT_PER_WAVE] = {};

    union Frag { v16bf v; u4 q[2]; };

    for (int kt = 0; kt < KSTEPS; ++kt) {
        // A fragment: e -> K = kt*32+hsel*16+e (matches packed B order)
        // -> elements 0..7 and 8..15 are each 8 contiguous bf16: 2 b128 LDS reads
        const int kA0 = kt * 32 + hsel * 16;
        Frag ah[MT], al[MT];
        #pragma unroll
        for (int mt = 0; mt < MT; ++mt) {
            ah[mt].q[0] = *(const u4*)&AsHi[mt][m][kA0];
            ah[mt].q[1] = *(const u4*)&AsHi[mt][m][kA0 + 8];
            al[mt].q[0] = *(const u4*)&AsLo[mt][m][kA0];
            al[mt].q[1] = *(const u4*)&AsLo[mt][m][kA0 + 8];
        }

        const size_t fbaseK = (((size_t)i * KSTEPS + kt) * CTILES);
        #pragma unroll
        for (int cc = 0; cc < CT_PER_WAVE; ++cc) {
            const int ct = wave * CT_PER_WAVE + cc;
            const size_t fo = ((fbaseK + ct) * 32 + lane) * 16;   // shorts
            Frag bhi, blo;
            bhi.q[0] = ((const u4*)(whi + fo))[0];
            bhi.q[1] = ((const u4*)(whi + fo))[1];
            blo.q[0] = ((const u4*)(wlo + fo))[0];
            blo.q[1] = ((const u4*)(wlo + fo))[1];
            // each 64B B fragment feeds 4 M-tiles x 3 split products = 12 WMMAs
            #pragma unroll
            for (int mt = 0; mt < MT; ++mt) {
                acc[mt][cc] = __builtin_amdgcn_wmma_f32_16x16x32_bf16(
                    false, ah[mt].v, false, bhi.v, (short)0, acc[mt][cc], false, false);
            }
            #pragma unroll
            for (int mt = 0; mt < MT; ++mt) {
                acc[mt][cc] = __builtin_amdgcn_wmma_f32_16x16x32_bf16(
                    false, ah[mt].v, false, blo.v, (short)0, acc[mt][cc], false, false);
            }
            #pragma unroll
            for (int mt = 0; mt < MT; ++mt) {
                acc[mt][cc] = __builtin_amdgcn_wmma_f32_16x16x32_bf16(
                    false, al[mt].v, false, bhi.v, (short)0, acc[mt][cc], false, false);
            }
        }
    }

    // ---- scatter-add (scale already folded into A) ----
    // C layout: VGPR r holds row M = r + 8*hsel, column N = (lane&15) within tile
    int rows[8];
    #pragma unroll
    for (int r = 0; r < 8; ++r) rows[r] = sIdx[r + 8 * hsel];
    #pragma unroll
    for (int mt = 0; mt < MT; ++mt) {
        float* nb = next + (size_t)(b0 + mt) * NROWS * DDIM;
        #pragma unroll
        for (int cc = 0; cc < CT_PER_WAVE; ++cc) {
            const int n = (wave * CT_PER_WAVE + cc) * 16 + m;
            #pragma unroll
            for (int r = 0; r < 8; ++r) {
                unsafeAtomicAdd(nb + (size_t)rows[r] * DDIM + n, acc[mt][cc][r]);
            }
        }
    }
}

// ---------------------------------------------------------------------------
extern "C" void kernel_launch(void* const* d_in, const int* in_sizes, int n_in,
                              void* d_out, int out_size, void* d_ws, size_t ws_size,
                              hipStream_t stream) {
    const float* x      = (const float*)d_in[0];
    const float* W      = (const float*)d_in[1];
    const int*   groups = (const int*)d_in[2];
    float* out = (float*)d_out;

    const size_t NSTATE = (size_t)BATCH * NROWS * DDIM;   // 33.5M elems
    char* ws = (char*)d_ws;
    float*          buf = (float*)ws;                               // 134 MB ping buffer
    unsigned short* whi = (unsigned short*)(ws + NSTATE * 4);       // 2 MB
    unsigned short* wlo = whi + (size_t)NPROJ * DDIM * DDIM;        // 2 MB
    unsigned short* xhi = wlo + (size_t)NPROJ * DDIM * DDIM;        // 67 MB (per-iter split)
    unsigned short* xlo = xhi + NSTATE;                             // 67 MB

    // split W into bf16 hi/lo WMMA fragments (once per call)
    pack_w_kernel<<<(NPROJ * KSTEPS * CTILES * 32) / 256, 256, 0, stream>>>(W, whi, wlo);

    const int n4 = (int)(NSTATE / 4);
    const int cpBlocks = (n4 + 255) / 256;
    dim3 ggrid(NGRP, BATCH / MT, NPROJ);

    // iter 1: cur = x   -> next = out   (scale = 1)
    split_copy_kernel<<<cpBlocks, 256, 0, stream>>>((const f4*)x, (f4*)out, xhi, xlo, 1.0f, n4);
    group_gemm_kernel<<<ggrid, 128, 0, stream>>>(xhi, xlo, out, whi, wlo, groups);

    // iter 2: cur = out -> next = buf   (scale = 1/2)
    split_copy_kernel<<<cpBlocks, 256, 0, stream>>>((const f4*)out, (f4*)buf, xhi, xlo, 0.5f, n4);
    group_gemm_kernel<<<ggrid, 128, 0, stream>>>(xhi, xlo, buf, whi, wlo, groups);

    // iter 3: cur = buf -> next = out   (scale = 1/3)
    split_copy_kernel<<<cpBlocks, 256, 0, stream>>>((const f4*)buf, (f4*)out, xhi, xlo, 1.0f / 3.0f, n4);
    group_gemm_kernel<<<ggrid, 128, 0, stream>>>(xhi, xlo, out, whi, wlo, groups);
}
